// RNNNetwork_34900904247946
// MI455X (gfx1250) — matrix operand order repaired and checked by reading, not compiled, not run on hardware
//
#include <hip/hip_runtime.h>
#include <hip/hip_bf16.h>

// ---------------------------------------------------------------------------
// 2-layer LSTM (T=512,B=128,IN=64,H=512,OUT=64) for gfx1250 (MI455X).
//   * convert x / weights to bf16
//   * xproj (both layers) + final linear as big WMMA bf16 GEMMs (parallel in T)
//   * recurrence: 16-block kernel, grid barrier per step, double-buffered h,
//     c-state in registers, Wh slice staged in LDS (128KB of the 320KB WGP LDS)
//     via the Tensor Data Mover (TDM) when available.
// ---------------------------------------------------------------------------

typedef __attribute__((ext_vector_type(16))) __bf16 vbf16x16;
typedef __attribute__((ext_vector_type(8)))  float  vf32x8;
typedef __attribute__((ext_vector_type(4)))  unsigned int v4u;
typedef __attribute__((ext_vector_type(8)))  int v8i;
typedef __attribute__((ext_vector_type(4)))  int v4i;

union Frag {
    vbf16x16 v;
    uint4    q[2];
};

#define Tn   512
#define Bn   128
#define INn  64
#define Hn   512
#define OUTn 64
#define G4H  2048          // 4*H
#define TBn  65536         // T*B

#if defined(__has_builtin)
#if __has_builtin(__builtin_amdgcn_tensor_load_to_lds) && \
    __has_builtin(__builtin_amdgcn_s_wait_tensorcnt)
#define HAVE_TDM 1
#endif
#endif

// ---------------------------------------------------------------------------
__global__ void f32_to_bf16_kernel(const float* __restrict__ in,
                                   __bf16* __restrict__ out, int n) {
    int i = blockIdx.x * blockDim.x + threadIdx.x;
    if (i < n) out[i] = (__bf16)in[i];
}

__global__ void bias_add_kernel(const float* __restrict__ a,
                                const float* __restrict__ b,
                                float* __restrict__ o, int n) {
    int i = blockIdx.x * blockDim.x + threadIdx.x;
    if (i < n) o[i] = a[i] + b[i];
}

// ---------------------------------------------------------------------------
// C[M,N] = A[M,K] * W[N,K]^T (+bias), bf16 in, fp32 accum/out.
// Block: 256 threads = 8 waves; 128 rows x (16*NT) cols per block.
template <int NT>
__global__ __launch_bounds__(256)
void gemm_bf16_kernel(const __bf16* __restrict__ A,
                      const __bf16* __restrict__ W,
                      const float*  __restrict__ bias,
                      float* __restrict__ C,
                      int M, int N, int K) {
    const int wave = threadIdx.x >> 5;
    const int lane = threadIdx.x & 31;
    const int lm   = lane & 15;
    const int hs   = (lane < 16) ? 0 : 1;
    const int mBase = blockIdx.y * 128 + wave * 16;
    const int nBase = blockIdx.x * (16 * NT);

    vf32x8 acc[NT] = {};

    for (int k = 0; k < K; k += 32) {
        Frag a;
        const __bf16* ap = A + (size_t)(mBase + lm) * K + k + hs * 8;
        a.q[0] = *(const uint4*)ap;
        a.q[1] = *(const uint4*)(ap + 16);
        __builtin_prefetch((const void*)(ap + 32), 0, 1);  // global_prefetch_b8
#pragma unroll
        for (int nt = 0; nt < NT; ++nt) {
            Frag bf;
            const __bf16* wp = W + (size_t)(nBase + nt * 16 + lm) * K + k + hs * 8;
            bf.q[0] = *(const uint4*)wp;
            bf.q[1] = *(const uint4*)(wp + 16);
            acc[nt] = __builtin_amdgcn_wmma_f32_16x16x32_bf16(
                false, a.v, false, bf.v, (short)0, acc[nt], false, false);
        }
    }

#pragma unroll
    for (int nt = 0; nt < NT; ++nt) {
        int col = nBase + nt * 16 + lm;
        float bv = bias ? bias[col] : 0.0f;
#pragma unroll
        for (int r = 0; r < 8; ++r) {
            int m = mBase + r + hs * 8;
            C[(size_t)m * N + col] = acc[nt][r] + bv;
        }
    }
}

// ---------------------------------------------------------------------------
// LSTM recurrence. grid = 16 blocks x 256 threads. Block b owns hidden units
// n in [32b,32b+32) for all 4 gates. Wave w owns rows m in [16w,16w+16).
// Block's Wh slice (4 gate-blocks of 32 contiguous rows x 512 K bf16 = 128KB)
// is DMA'd into LDS by the TDM, then stays resident for all 512 steps.
__global__ __launch_bounds__(256)
void lstm_rec_kernel(const __bf16* __restrict__ Wh,     // [2048,512] gate-major
                     const float*  __restrict__ bcomb,  // [2048] bx+bh
                     const float*  __restrict__ xp,     // [T,128,2048]
                     const float*  __restrict__ c0,     // [128,512]
                     __bf16* __restrict__ hbuf,         // [2,128,512], [0]=h0
                     __bf16* __restrict__ out,          // [T,128,512]
                     float* __restrict__ hT,            // [128,512] -> d_out
                     float* __restrict__ cT,            // [128,512] -> d_out
                     unsigned int* __restrict__ bar) {
    // local row (g*32 + j)  <->  global Wh row (g*512 + b*32 + j), j in [0,32)
    __shared__ __bf16 whs[128 * Hn];    // 128 KB of the 320 KB WGP LDS

    const int b    = blockIdx.x;
    const int wave = threadIdx.x >> 5;
    const int lane = threadIdx.x & 31;
    const int lm   = lane & 15;
    const int hs   = (lane < 16) ? 0 : 1;
    const int mA   = wave * 16 + lm;  // row for A-fragment loads

#ifdef HAVE_TDM
    // ---- stage Wh slice into LDS with the Tensor Data Mover ----
    // TDM issues per-wave and ignores EXEC, so only wave 0 may reach it.
    if (threadIdx.x < 32) {
        for (int g = 0; g < 4; ++g) {
            unsigned lds_addr = (unsigned)(size_t)(&whs[g * 32 * Hn]);
            unsigned long long ga =
                (unsigned long long)(size_t)(Wh + ((size_t)g * Hn + (size_t)b * 32) * Hn);
            v4u g0;
            g0[0] = 1u;                                   // count=1 user D#
            g0[1] = lds_addr;                             // LDS dest (bytes)
            g0[2] = (unsigned)(ga & 0xffffffffu);         // global_addr[31:0]
            g0[3] = (unsigned)((ga >> 32) & 0x1ffffffu)   // global_addr[56:32]
                    | (2u << 30);                         // type=2 (image)
            v8i g1;
            g1[0] = (int)(1u << 16);     // workgroup_mask=0, data_size=2B
            g1[1] = (int)(512u << 16);   // tensor_dim0 = 512 (low16)
            g1[2] = (int)(32u << 16);    // tensor_dim0 hi=0, tensor_dim1 = 32
            g1[3] = (int)(512u << 16);   // tensor_dim1 hi=0, tile_dim0 = 512
            g1[4] = 32;                  // tile_dim1 = 32, tile_dim2 = 0
            g1[5] = 512;                 // tensor_dim0_stride = 512 elems
            g1[6] = 0;
            g1[7] = 0;
            v4i z4 = {0, 0, 0, 0};       // groups 2/3 unused (2D tensor)
#if __clang_major__ >= 23
            v8i z8 = {0, 0, 0, 0, 0, 0, 0, 0};
            __builtin_amdgcn_tensor_load_to_lds(g0, g1, z4, z4, z8, 0);
#else
            __builtin_amdgcn_tensor_load_to_lds(g0, g1, z4, z4, 0);
#endif
        }
        __builtin_amdgcn_s_wait_tensorcnt(0);   // s_wait_tensorcnt 0
    }
#else
    // ---- fallback: cooperative copy (uint4 = 8 bf16 granules) ----
    for (int idx = threadIdx.x; idx < 128 * Hn / 8; idx += 256) {
        int lr  = idx >> 6;          // 64 uint4 per 512-elem row
        int cq  = idx & 63;
        int g   = lr >> 5;
        int j   = lr & 31;
        int rw  = g * Hn + b * 32 + j;
        ((uint4*)whs)[idx] = ((const uint4*)(Wh + (size_t)rw * Hn))[cq];
    }
#endif

    float creg[2][8];
    float breg[2][4];
#pragma unroll
    for (int tt = 0; tt < 2; ++tt) {
        int n = b * 32 + tt * 16 + lm;
#pragma unroll
        for (int g = 0; g < 4; ++g) breg[tt][g] = bcomb[g * Hn + n];
#pragma unroll
        for (int r = 0; r < 8; ++r) {
            int m = wave * 16 + r + hs * 8;
            creg[tt][r] = c0[m * Hn + n];
        }
    }
    __syncthreads();

    for (int t = 0; t < Tn; ++t) {
        // stop LICM from hoisting the (loop-invariant) LDS weight fragments
        // out of the t-loop -- that would spill ~1K VGPRs to scratch.
        asm volatile("" ::: "memory");

        const __bf16* hprev = hbuf + (size_t)(t & 1) * (Bn * Hn);
        __bf16* hnext       = hbuf + (size_t)((t + 1) & 1) * (Bn * Hn);

        vf32x8 acc[8] = {};
        for (int k = 0; k < Hn; k += 32) {
            Frag a;
            const __bf16* ap = hprev + (size_t)mA * Hn + k + hs * 8;
            a.q[0] = *(const uint4*)ap;
            a.q[1] = *(const uint4*)(ap + 16);
#pragma unroll
            for (int nt = 0; nt < 8; ++nt) {
                // acc tile nt -> gate nt>>1, sub-block tt=nt&1
                // local LDS row = (nt>>1)*32 + (nt&1)*16 + lm
                const __bf16* wp =
                    whs + ((nt >> 1) * 32 + (nt & 1) * 16 + lm) * Hn + k + hs * 8;
                Frag bf;
                bf.q[0] = *(const uint4*)wp;   // ds_load_b128
                bf.q[1] = *(const uint4*)(wp + 16);
                acc[nt] = __builtin_amdgcn_wmma_f32_16x16x32_bf16(
                    false, a.v, false, bf.v, (short)0, acc[nt], false, false);
            }
        }

        const float* xpt = xp + (size_t)t * (Bn * G4H);
#pragma unroll
        for (int tt = 0; tt < 2; ++tt) {
            int n = b * 32 + tt * 16 + lm;
#pragma unroll
            for (int r = 0; r < 8; ++r) {
                int m = wave * 16 + r + hs * 8;
                const float* xr = xpt + (size_t)m * G4H + n;
                float gi = acc[0 + tt][r] + xr[0 * Hn] + breg[tt][0];
                float gj = acc[2 + tt][r] + xr[1 * Hn] + breg[tt][1];
                float gf = acc[4 + tt][r] + xr[2 * Hn] + breg[tt][2];
                float go = acc[6 + tt][r] + xr[3 * Hn] + breg[tt][3];
                float ig = 1.0f / (1.0f + __expf(-gi));
                float jg = tanhf(gj);
                float fg = 1.0f / (1.0f + __expf(-gf));
                float og = 1.0f / (1.0f + __expf(-go));
                float cc = fg * creg[tt][r] + ig * jg;
                creg[tt][r] = cc;
                float hh = og * tanhf(cc);
                __bf16 hb = (__bf16)hh;
                hnext[(size_t)m * Hn + n] = hb;
                out[(size_t)t * (Bn * Hn) + (size_t)m * Hn + n] = hb;
                if (t == Tn - 1) {
                    hT[m * Hn + n] = hh;
                    cT[m * Hn + n] = cc;
                }
            }
        }

        // ---- device-wide barrier (16 resident blocks) ----
        __threadfence();
        __syncthreads();
        if (threadIdx.x == 0) {
            atomicAdd(bar, 1u);
            unsigned target = (unsigned)gridDim.x * (unsigned)(t + 1);
            volatile unsigned int* vb = bar;
            while (*vb < target) { __builtin_amdgcn_s_sleep(1); }
        }
        __syncthreads();
        __threadfence();
    }
}

// ---------------------------------------------------------------------------
extern "C" void kernel_launch(void* const* d_in, const int* in_sizes, int n_in,
                              void* d_out, int out_size, void* d_ws, size_t ws_size,
                              hipStream_t stream) {
    const float* x    = (const float*)d_in[0];   // [T,B,IN]
    const float* hid  = (const float*)d_in[1];   // [2,B,H]
    const float* cin  = (const float*)d_in[2];   // [2,B,H]
    const float* Wx0  = (const float*)d_in[3];   // [4,H,IN]
    const float* bx0  = (const float*)d_in[4];   // [4,H]
    const float* Wh0  = (const float*)d_in[5];   // [4,H,H]
    const float* bh0  = (const float*)d_in[6];   // [4,H]
    const float* Wx1  = (const float*)d_in[7];   // [4,H,H]
    const float* bx1  = (const float*)d_in[8];   // [4,H]
    const float* Wh1  = (const float*)d_in[9];   // [4,H,H]
    const float* bh1  = (const float*)d_in[10];  // [4,H]
    const float* Wf   = (const float*)d_in[11];  // [OUT,H]
    const float* bf   = (const float*)d_in[12];  // [OUT]
    float* out = (float*)d_out;                  // [T*B*OUT | hcat | ccat]

    // ---- workspace layout (256B aligned chunks) ----
    char* ws = (char*)d_ws;
    size_t off = 0;
    auto alloc = [&](size_t bytes) {
        char* p = ws + off;
        off += (bytes + 255) & ~(size_t)255;
        return p;
    };
    __bf16* xbf    = (__bf16*)alloc((size_t)TBn * INn * 2);
    __bf16* Wx0b   = (__bf16*)alloc((size_t)G4H * INn * 2);
    __bf16* Wh0b   = (__bf16*)alloc((size_t)G4H * Hn * 2);
    __bf16* Wx1b   = (__bf16*)alloc((size_t)G4H * Hn * 2);
    __bf16* Wh1b   = (__bf16*)alloc((size_t)G4H * Hn * 2);
    __bf16* Wfb    = (__bf16*)alloc((size_t)OUTn * Hn * 2);
    float*  bc0    = (float*)alloc((size_t)G4H * 4);
    float*  bc1    = (float*)alloc((size_t)G4H * 4);
    __bf16* hbuf0  = (__bf16*)alloc((size_t)2 * Bn * Hn * 2);
    __bf16* hbuf1  = (__bf16*)alloc((size_t)2 * Bn * Hn * 2);
    unsigned int* bars = (unsigned int*)alloc(256);
    __bf16* out0b  = (__bf16*)alloc((size_t)TBn * Hn * 2);
    __bf16* out1b  = (__bf16*)alloc((size_t)TBn * Hn * 2);
    float*  xp     = (float*)alloc((size_t)TBn * G4H * 4);

    // d_out sections
    float* hT0 = out + (size_t)TBn * OUTn;
    float* hT1 = hT0 + (size_t)Bn * Hn;
    float* cT0 = hT1 + (size_t)Bn * Hn;
    float* cT1 = cT0 + (size_t)Bn * Hn;

    hipMemsetAsync(bars, 0, 256, stream);

    auto cvt = [&](const float* src, __bf16* dst, int n) {
        f32_to_bf16_kernel<<<(n + 255) / 256, 256, 0, stream>>>(src, dst, n);
    };
    cvt(x,   xbf,  TBn * INn);
    cvt(Wx0, Wx0b, G4H * INn);
    cvt(Wh0, Wh0b, G4H * Hn);
    cvt(Wx1, Wx1b, G4H * Hn);
    cvt(Wh1, Wh1b, G4H * Hn);
    cvt(Wf,  Wfb,  OUTn * Hn);
    cvt(hid,            hbuf0, Bn * Hn);   // layer0 h0 -> buffer slot 0
    cvt(hid + Bn * Hn,  hbuf1, Bn * Hn);   // layer1 h0 -> buffer slot 0
    bias_add_kernel<<<(G4H + 255) / 256, 256, 0, stream>>>(bx0, bh0, bc0, G4H);
    bias_add_kernel<<<(G4H + 255) / 256, 256, 0, stream>>>(bx1, bh1, bc1, G4H);

    // xproj layer 0: [65536,64] x [2048,64]^T -> xp
    {
        dim3 g(G4H / 128, TBn / 128);
        gemm_bf16_kernel<8><<<g, 256, 0, stream>>>(xbf, Wx0b, nullptr, xp,
                                                   TBn, G4H, INn);
    }
    // recurrence layer 0
    lstm_rec_kernel<<<16, 256, 0, stream>>>(Wh0b, bc0, xp, cin,
                                            hbuf0, out0b, hT0, cT0, bars + 0);
    // xproj layer 1: [65536,512] x [2048,512]^T -> xp (reuse)
    {
        dim3 g(G4H / 128, TBn / 128);
        gemm_bf16_kernel<8><<<g, 256, 0, stream>>>(out0b, Wx1b, nullptr, xp,
                                                   TBn, G4H, Hn);
    }
    // recurrence layer 1
    lstm_rec_kernel<<<16, 256, 0, stream>>>(Wh1b, bc1, xp, cin + (size_t)Bn * Hn,
                                            hbuf1, out1b, hT1, cT1, bars + 32);
    // final linear: [65536,512] x [64,512]^T + bias -> d_out
    {
        dim3 g(OUTn / 64, TBn / 128);
        gemm_bf16_kernel<4><<<g, 256, 0, stream>>>(out1b, Wfb, bf, out,
                                                   TBn, OUTn, Hn);
    }
    (void)in_sizes; (void)n_in; (void)out_size; (void)ws_size;
}
